// GGNN_14199161880902
// MI455X (gfx1250) — matrix operand order repaired and checked by reading, not compiled, not run on hardware
//
#include <hip/hip_runtime.h>
#include <hip/hip_bf16.h>
#include <math.h>

typedef __attribute__((ext_vector_type(16))) _Float16 v16h;
typedef __attribute__((ext_vector_type(8)))  _Float16 v8h;
typedef __attribute__((ext_vector_type(8)))  float    v8f;

constexpr int kH    = 150;   // hidden size
constexpr int kHP   = 160;   // K padded to multiple of 32
constexpr int kG3   = 450;   // 3*H gate columns
constexpr int kG3P  = 464;   // padded to 29*16
constexpr int kColT = 29;    // number of 16-wide column tiles
constexpr int kMT   = 32;    // nodes per block (M tile = two 16-row WMMA tiles)

// ---------------------------------------------------------------------------
// Convert GRU weights to f16, padded layout [464 rows][160 cols], row-major.
// Row = gate output column, col = K. Zero padding beyond 450 / 150.
// ---------------------------------------------------------------------------
__global__ void convert_weights(const float* __restrict__ wih,
                                const float* __restrict__ whh,
                                _Float16* __restrict__ wih16,
                                _Float16* __restrict__ whh16) {
  int idx = blockIdx.x * blockDim.x + threadIdx.x;
  if (idx >= kG3P * kHP) return;
  int col = idx / kHP, k = idx % kHP;
  float a = 0.f, b = 0.f;
  if (col < kG3 && k < kH) { a = wih[col * kH + k]; b = whh[col * kH + k]; }
  wih16[idx] = (_Float16)a;
  whh16[idx] = (_Float16)b;
}

// ---------------------------------------------------------------------------
// incoming[dst] += h[src] : one wave per edge, lanes stride features.
// ---------------------------------------------------------------------------
__global__ void edge_scatter(const int* __restrict__ src,
                             const int* __restrict__ dst,
                             const float* __restrict__ h,
                             float* __restrict__ incoming, int nEdges) {
  int wave = blockIdx.x * (blockDim.x >> 5) + (threadIdx.x >> 5);
  int lane = threadIdx.x & 31;
  if (wave >= nEdges) return;
  int s = src[wave], d = dst[wave];
  const float* hs = h + (size_t)s * kH;
  float* out = incoming + (size_t)d * kH;
  for (int f = lane; f < kH; f += 32) atomicAdd(&out[f], hs[f]);
}

// ---------------------------------------------------------------------------
// Fused GRU cell: gi = x@Wih^T, gh = h@Whh^T via WMMA f16 (fp32 acc).
// 32-node M-tile per block: each wave computes TWO 16-row accumulator pairs
// per column tile so every B (weight) fragment is reused 2x, halving the
// dominant L2 weight stream. h updated in place (block owns its 32 rows).
// ---------------------------------------------------------------------------
__global__ void __launch_bounds__(256)
gru_wmma(const float* __restrict__ incoming, float* __restrict__ h,
         const _Float16* __restrict__ wih16, const _Float16* __restrict__ whh16,
         const float* __restrict__ b_ih, const float* __restrict__ b_hh,
         int nNodes) {
  __shared__ __align__(16) _Float16 sh_x[kMT][kHP];
  __shared__ __align__(16) _Float16 sh_h[kMT][kHP];
  __shared__ __align__(16) float sh_rz[kMT][304];   // gi+gh for cols 0..299
  __shared__ __align__(16) float sh_in[kMT][kHP];   // gi for n-gate cols
  __shared__ __align__(16) float sh_hn[kMT][kHP];   // gh for n-gate cols

  const int m0 = blockIdx.x * kMT;

  // cooperative load + f16 convert of the 32-node tile (K padded with zeros)
  for (int idx = threadIdx.x; idx < kMT * kHP; idx += blockDim.x) {
    int r = idx / kHP, c = idx % kHP;
    int node = m0 + r;
    float xv = 0.f, hv = 0.f;
    if (node < nNodes && c < kH) {
      xv = incoming[(size_t)node * kH + c];
      hv = h[(size_t)node * kH + c];
    }
    sh_x[r][c] = (_Float16)xv;
    sh_h[r][c] = (_Float16)hv;
  }
  __syncthreads();

  const int lane = threadIdx.x & 31;
  const int wave = threadIdx.x >> 5;
  const int half = lane >> 4;       // which half-wave
  const int nidx = lane & 15;       // output column within tile (B/D layout)
  const int m    = lane & 15;       // row for A layout

  for (int c = wave; c < kColT; c += 8) {
    v8f acc_i0 = {}, acc_h0 = {};   // rows 0..15
    v8f acc_i1 = {}, acc_h1 = {};   // rows 16..31
    const _Float16* bi_base = wih16 + (size_t)(c * 16 + nidx) * kHP + half * 16;
    const _Float16* bh_base = whh16 + (size_t)(c * 16 + nidx) * kHP + half * 16;
#pragma unroll
    for (int kt = 0; kt < kHP / 32; kt++) {
      // B 32x16 f16 operand, k order chosen here: lane = column, 16 contiguous K per half-wave
      v16h bi = *(const v16h*)(bi_base + kt * 32);
      v16h bh = *(const v16h*)(bh_base + kt * 32);
      // A 16x32 f16 operand, same chosen k order: lanes 0-15 hold K 0-15; lanes 16-31 K 16-31 (matches B)
      v8h x0lo = *(const v8h*)&sh_x[m][kt * 32 + half * 16];
      v8h x0hi = *(const v8h*)&sh_x[m][kt * 32 + 8 + half * 16];
      v8h h0lo = *(const v8h*)&sh_h[m][kt * 32 + half * 16];
      v8h h0hi = *(const v8h*)&sh_h[m][kt * 32 + 8 + half * 16];
      v8h x1lo = *(const v8h*)&sh_x[m + 16][kt * 32 + half * 16];
      v8h x1hi = *(const v8h*)&sh_x[m + 16][kt * 32 + 8 + half * 16];
      v8h h1lo = *(const v8h*)&sh_h[m + 16][kt * 32 + half * 16];
      v8h h1hi = *(const v8h*)&sh_h[m + 16][kt * 32 + 8 + half * 16];
      v16h ax0, ah0, ax1, ah1;
#pragma unroll
      for (int e = 0; e < 8; e++) {
        ax0[e] = x0lo[e]; ax0[e + 8] = x0hi[e];
        ah0[e] = h0lo[e]; ah0[e + 8] = h0hi[e];
        ax1[e] = x1lo[e]; ax1[e + 8] = x1hi[e];
        ah1[e] = h1lo[e]; ah1[e + 8] = h1hi[e];
      }
      acc_i0 = __builtin_amdgcn_wmma_f32_16x16x32_f16(false, ax0, false, bi,
                                                      (short)0, acc_i0, false, false);
      acc_i1 = __builtin_amdgcn_wmma_f32_16x16x32_f16(false, ax1, false, bi,
                                                      (short)0, acc_i1, false, false);
      acc_h0 = __builtin_amdgcn_wmma_f32_16x16x32_f16(false, ah0, false, bh,
                                                      (short)0, acc_h0, false, false);
      acc_h1 = __builtin_amdgcn_wmma_f32_16x16x32_f16(false, ah1, false, bh,
                                                      (short)0, acc_h1, false, false);
    }
    // D layout: lane -> N = lane&15, element v -> M = 8*(lane>>4)+v
    int C = c * 16 + nidx;
#pragma unroll
    for (int v = 0; v < 8; v++) {
      int row = half * 8 + v;
      if (C < 2 * kH) {
        sh_rz[row][C]      = acc_i0[v] + acc_h0[v];    // r,z only need the sum
        sh_rz[row + 16][C] = acc_i1[v] + acc_h1[v];
      } else if (C < 3 * kH) {
        sh_in[row][C - 2 * kH]      = acc_i0[v];       // n gate needs both
        sh_hn[row][C - 2 * kH]      = acc_h0[v];
        sh_in[row + 16][C - 2 * kH] = acc_i1[v];
        sh_hn[row + 16][C - 2 * kH] = acc_h1[v];
      }
    }
  }
  __syncthreads();

  // elementwise GRU update, in place
  for (int idx = threadIdx.x; idx < kMT * kH; idx += blockDim.x) {
    int r = idx / kH, j = idx % kH;
    int node = m0 + r;
    if (node >= nNodes) continue;
    float rr = sh_rz[r][j] + b_ih[j] + b_hh[j];
    rr = 1.f / (1.f + expf(-rr));
    float zz = sh_rz[r][kH + j] + b_ih[kH + j] + b_hh[kH + j];
    zz = 1.f / (1.f + expf(-zz));
    float nn = sh_in[r][j] + b_ih[2 * kH + j] + rr * (sh_hn[r][j] + b_hh[2 * kH + j]);
    nn = tanhf(nn);
    float hold = h[(size_t)node * kH + j];
    h[(size_t)node * kH + j] = (1.f - zz) * nn + zz * hold;
  }
}

// ---------------------------------------------------------------------------
// Per-graph readout
// ---------------------------------------------------------------------------
__global__ void graph_scatter(const int* __restrict__ gid,
                              const float* __restrict__ h,
                              float* __restrict__ gsum, int nNodes) {
  int wave = blockIdx.x * (blockDim.x >> 5) + (threadIdx.x >> 5);
  int lane = threadIdx.x & 31;
  if (wave >= nNodes) return;
  int g = gid[wave];
  const float* hs = h + (size_t)wave * kH;
  float* out = gsum + (size_t)g * kH;
  for (int f = lane; f < kH; f += 32) atomicAdd(&out[f], hs[f]);
}

__global__ void log_nan_relu(float* g, int n) {
  int idx = blockIdx.x * blockDim.x + threadIdx.x;
  if (idx >= n) return;
  float v = logf(g[idx]);
  if (v != v) v = 0.f;      // nan -> 0
  v = fmaxf(v, 0.f);        // relu (relu(-inf) = 0)
  g[idx] = v;
}

__global__ void fc_kernel(const float* __restrict__ in, const float* __restrict__ w,
                          const float* __restrict__ b, float* __restrict__ out,
                          int G, int IN, int OUT, int leaky) {
  int idx = blockIdx.x * blockDim.x + threadIdx.x;
  if (idx >= G * OUT) return;
  int g = idx / OUT, o = idx % OUT;
  float acc = b[o];
  for (int k = 0; k < IN; k++) acc += in[g * IN + k] * w[o * IN + k];
  if (leaky) acc = acc > 0.f ? acc : 0.01f * acc;
  out[idx] = acc;
}

// ---------------------------------------------------------------------------
extern "C" void kernel_launch(void* const* d_in, const int* in_sizes, int n_in,
                              void* d_out, int out_size, void* d_ws, size_t ws_size,
                              hipStream_t stream) {
  const float* nodes = (const float*)d_in[0];
  const int*   src   = (const int*)d_in[1];
  const int*   dst   = (const int*)d_in[2];
  const int*   gid   = (const int*)d_in[3];
  const float* w_ih  = (const float*)d_in[4];
  const float* w_hh  = (const float*)d_in[5];
  const float* b_ih  = (const float*)d_in[6];
  const float* b_hh  = (const float*)d_in[7];
  const float* fc1_w = (const float*)d_in[8];
  const float* fc1_b = (const float*)d_in[9];
  const float* fc2_w = (const float*)d_in[10];
  const float* fc2_b = (const float*)d_in[11];
  const float* fcl_w = (const float*)d_in[12];
  const float* fcl_b = (const float*)d_in[13];

  const int nNodes  = in_sizes[0] / kH;   // 100000
  const int nEdges  = in_sizes[1];        // 1600000
  const int nGraphs = out_size / 10;      // 64

  char* ws = (char*)d_ws;
  size_t off = 0;
  auto alloc = [&](size_t bytes) -> char* {
    char* p = ws + off;
    off += (bytes + 255) & ~(size_t)255;
    return p;
  };
  _Float16* wih16    = (_Float16*)alloc((size_t)kG3P * kHP * sizeof(_Float16));
  _Float16* whh16    = (_Float16*)alloc((size_t)kG3P * kHP * sizeof(_Float16));
  float*    gsum     = (float*)alloc((size_t)nGraphs * kH * sizeof(float));
  float*    x1       = (float*)alloc((size_t)nGraphs * 80 * sizeof(float));
  float*    x2       = (float*)alloc((size_t)nGraphs * 80 * sizeof(float));
  float*    incoming = (float*)alloc((size_t)nNodes * kH * sizeof(float));
  float*    hbuf     = (float*)alloc((size_t)nNodes * kH * sizeof(float));

  // per-call weight conversion + h init (deterministic, capture-safe)
  {
    int n = kG3P * kHP;
    convert_weights<<<(n + 255) / 256, 256, 0, stream>>>(w_ih, w_hh, wih16, whh16);
  }
  hipMemcpyAsync(hbuf, nodes, (size_t)nNodes * kH * sizeof(float),
                 hipMemcpyDeviceToDevice, stream);

  const int edgeBlocks = (nEdges + 7) / 8;              // 8 waves / block
  const int gruBlocks  = (nNodes + kMT - 1) / kMT;      // 3125

  for (int p = 0; p < 4; p++) {
    hipMemsetAsync(incoming, 0, (size_t)nNodes * kH * sizeof(float), stream);
    edge_scatter<<<edgeBlocks, 256, 0, stream>>>(src, dst, hbuf, incoming, nEdges);
    gru_wmma<<<gruBlocks, 256, 0, stream>>>(incoming, hbuf, wih16, whh16,
                                            b_ih, b_hh, nNodes);
  }

  hipMemsetAsync(gsum, 0, (size_t)nGraphs * kH * sizeof(float), stream);
  graph_scatter<<<(nNodes + 7) / 8, 256, 0, stream>>>(gid, hbuf, gsum, nNodes);
  log_nan_relu<<<(nGraphs * kH + 255) / 256, 256, 0, stream>>>(gsum, nGraphs * kH);
  fc_kernel<<<(nGraphs * 80 + 255) / 256, 256, 0, stream>>>(gsum, fc1_w, fc1_b, x1,
                                                            nGraphs, kH, 80, 1);
  fc_kernel<<<(nGraphs * 80 + 255) / 256, 256, 0, stream>>>(x1, fc2_w, fc2_b, x2,
                                                            nGraphs, 80, 80, 1);
  fc_kernel<<<(nGraphs * 10 + 255) / 256, 256, 0, stream>>>(x2, fcl_w, fcl_b,
                                                            (float*)d_out,
                                                            nGraphs, 80, 10, 0);
}
